// RoPESelfKANtentionND_3358664425751
// MI455X (gfx1250) — compile-verified, matrix operand order checked by reading, not run on hardware
//
#include <hip/hip_runtime.h>

#define B_ 64
#define C_ 256
#define N_ 1025
#define NP 1056          // padded token count: 33 chunks of 32, 66 tiles of 16
#define NEG_INF (-__builtin_inff())

typedef __attribute__((ext_vector_type(16))) __bf16       v16bf;
typedef __attribute__((ext_vector_type(4)))  __bf16       bf16x4;
typedef __attribute__((ext_vector_type(8)))  float        v8f;
typedef __attribute__((ext_vector_type(4)))  unsigned int u32x4;
typedef __attribute__((ext_vector_type(4)))  float        f32x4;

union AFrag {
    v16bf  v;
    bf16x4 b4[4];
    u32x4  q[2];
    unsigned short h[16];
};

__device__ __forceinline__ bf16x4 cvt4(f32x4 v) {
    return __builtin_convertvector(v, bf16x4);
}
__device__ __forceinline__ unsigned short cvt1(float f) {
    return __builtin_bit_cast(unsigned short, (__bf16)f);
}

__device__ __forceinline__ v8f wmma_bf16(const AFrag& a, const AFrag& b, v8f c) {
    return __builtin_amdgcn_wmma_f32_16x16x32_bf16(
        false, a.v, false, b.v, (short)0, c, false, false);
}

// ---------------------------------------------------------------------------
// Kernel 0: RoPE angle tables  cosT/sinT[t][p], t in [0,1024), p in [0,128)
// ---------------------------------------------------------------------------
__global__ void k_tables(const float* __restrict__ freqs,
                         float* __restrict__ cosT, float* __restrict__ sinT) {
    int idx = blockIdx.x * blockDim.x + threadIdx.x;
    if (idx >= 1024 * 128) return;
    int t = idx >> 7, p = idx & 127;
    float tx = (float)(t & 31);      // t % 32
    float ty = (float)(t >> 5);      // t / 32
    float ang = tx * freqs[p] + ty * freqs[128 + p];
    cosT[idx] = __cosf(ang);
    sinT[idx] = __sinf(ang);
}

// ---------------------------------------------------------------------------
// Kernel 1 helpers (forceinline, pointers passed directly so address-space
// inference keeps every access on the GLOBAL path, not FLAT).
// ---------------------------------------------------------------------------
__device__ __forceinline__ v8f gemm_tile(const float* __restrict__ W,
                                         const float* __restrict__ bias,
                                         const AFrag (&xfr)[8],
                                         int dt, int ln, int coff, int half) {
    v8f acc;
    #pragma unroll
    for (int r = 0; r < 8; ++r) acc[r] = bias[dt * 16 + half * 8 + r];
    #pragma unroll
    for (int ct = 0; ct < 8; ++ct) {       // K = 256 in chunks of 32
        AFrag a;
        const float* wr = W + (size_t)(dt * 16 + ln) * C_ + ct * 32 + coff;
        a.b4[0] = cvt4(*(const f32x4*)(wr));
        a.b4[1] = cvt4(*(const f32x4*)(wr + 4));
        a.b4[2] = cvt4(*(const f32x4*)(wr + 16));
        a.b4[3] = cvt4(*(const f32x4*)(wr + 20));
        acc = wmma_bf16(a, xfr[ct], acc);
    }
    return acc;
}

__device__ __forceinline__ void qk_epilogue(v8f acc,
                                            const float* __restrict__ cosT,
                                            const float* __restrict__ sinT,
                                            unsigned short* __restrict__ dst,
                                            int b, int n, int dt, int half) {
    // Fused RoPE: VGPR pair (2j, 2j+1) is channel pair
    // (dt*16 + 2j + 8*half, +1); table index p = dt*8 + half*4 + j.
    if (n >= 1 && n <= N_ - 1) {
        const int t = n - 1;
        const f32x4 cs = *(const f32x4*)(cosT + (size_t)t * 128 + dt * 8 + half * 4);
        const f32x4 sn = *(const f32x4*)(sinT + (size_t)t * 128 + dt * 8 + half * 4);
        #pragma unroll
        for (int j = 0; j < 4; ++j) {
            float e = acc[2 * j], o = acc[2 * j + 1];
            acc[2 * j]     = e * cs[j] - o * sn[j];
            acc[2 * j + 1] = e * sn[j] + o * cs[j];
        }
    }
    // Transposed store: lane owns (n, d0+half*8 .. +7) contiguous bf16.
    union { bf16x4 b4[2]; u32x4 u; } pk;
    pk.b4[0] = cvt4(__builtin_shufflevector(acc, acc, 0, 1, 2, 3));
    pk.b4[1] = cvt4(__builtin_shufflevector(acc, acc, 4, 5, 6, 7));
    *(u32x4*)&dst[(size_t)b * NP * C_ + (size_t)n * C_ + dt * 16 + half * 8] = pk.u;
}

// ---------------------------------------------------------------------------
// Kernel 1: QKV projection (WMMA bf16) + fused RoPE.
// Outputs: qT,kT as (B, NP, C) bf16 row-major; v as (B, C, NP) bf16 row-major.
// Block = 128 threads (4 waves); grid = (66 n-tiles, 64 batches).
// ---------------------------------------------------------------------------
__global__ __launch_bounds__(128, 1) void k_proj(
    const float* __restrict__ x,
    const float* __restrict__ Wq, const float* __restrict__ bq,
    const float* __restrict__ Wk, const float* __restrict__ bk,
    const float* __restrict__ Wv, const float* __restrict__ bv,
    const float* __restrict__ cosT, const float* __restrict__ sinT,
    unsigned short* __restrict__ qT, unsigned short* __restrict__ kT,
    unsigned short* __restrict__ vO)
{
    __shared__ unsigned short xs[16 * 264];   // (n, c) bf16, row stride 264
    const int nt  = blockIdx.x;               // 0..65
    const int b   = blockIdx.y;
    const int n0  = nt * 16;
    const int tid = threadIdx.x;

    // Stage x[b, :, n0..n0+15] into LDS transposed to (n, c) bf16, zero-padded.
    for (int rr = 0; rr < 2; ++rr) {
        int c = rr * 128 + tid;
        const float* xp = x + (size_t)b * C_ * N_ + (size_t)c * N_ + n0;
        #pragma unroll
        for (int i = 0; i < 16; ++i) {
            float v = (n0 + i < N_) ? xp[i] : 0.0f;
            xs[i * 264 + c] = cvt1(v);
        }
    }
    __syncthreads();

    const int lane = tid & 31;
    const int wv   = tid >> 5;
    const int half = lane >> 4;
    const int ln   = lane & 15;
    const int coff = half ? 8 : 0;

    // B-fragments depend only on the lane: load the 8 LDS fragments ONCE and
    // reuse across all 3 matrices x 4 d-tiles handled by this wave.
    AFrag xfr[8];
    #pragma unroll
    for (int ct = 0; ct < 8; ++ct) {
        const int base = ln * 264 + ct * 32 + coff;
        xfr[ct].q[0] = *(const u32x4*)&xs[base];
        xfr[ct].q[1] = *(const u32x4*)&xs[base + 16];
    }

    for (int di = 0; di < 4; ++di) {
        const int dt = wv * 4 + di;            // d-tile 0..15
        const int n  = n0 + ln;

        v8f acc = gemm_tile(Wq, bq, xfr, dt, ln, coff, half);
        qk_epilogue(acc, cosT, sinT, qT, b, n, dt, half);

        acc = gemm_tile(Wk, bk, xfr, dt, ln, coff, half);
        qk_epilogue(acc, cosT, sinT, kT, b, n, dt, half);

        acc = gemm_tile(Wv, bv, xfr, dt, ln, coff, half);
        // v stays (C, NP): 8 scalar bf16 stores (rows differ).
        #pragma unroll
        for (int r = 0; r < 8; ++r) {
            const int d = dt * 16 + r + 8 * half;
            vO[(size_t)b * C_ * NP + (size_t)d * NP + n] = cvt1(acc[r]);
        }
    }
}

// ---------------------------------------------------------------------------
// Kernel 2: attention. One wave per 16-query tile; 4 waves/block; each wave
// owns a 16 x 1056 f32 energy slab in LDS (270336 B per block total).
// Writes y = gamma*x + attn_out into yW (B, C, NP) f32.
// ---------------------------------------------------------------------------
__global__ __launch_bounds__(128, 1) void k_attn(
    const unsigned short* __restrict__ qT,
    const unsigned short* __restrict__ kT,
    const unsigned short* __restrict__ vB,
    const float* __restrict__ x,
    const float* __restrict__ gamma,
    float* __restrict__ yW)
{
    extern __shared__ char smem[];
    const int b    = blockIdx.x;
    const int tid  = threadIdx.x;
    const int wv   = tid >> 5;
    const int lane = tid & 31;
    const int qt   = blockIdx.y * 4 + wv;
    if (qt > 64) return;                        // 65 query tiles cover N=1025
    const int q0   = qt * 16;
    const int half = lane >> 4;
    const int ln   = lane & 15;
    const int coff = half ? 8 : 0;

    float* E = (float*)smem + (size_t)wv * 16 * NP;

    // Preload the 16x256 query panel as 8 A-fragments (kept in registers).
    AFrag qf[8];
    {
        const unsigned short* qb = qT + (size_t)b * NP * C_ + (size_t)(q0 + ln) * C_;
        #pragma unroll
        for (int ct = 0; ct < 8; ++ct) {
            const unsigned short* p = qb + ct * 32 + coff;
            qf[ct].q[0] = *(const u32x4*)p;
            qf[ct].q[1] = *(const u32x4*)(p + 16);
        }
    }

    // ---- Phase 1: S = q . k^T into LDS, masking cols >= N_ ----
    const unsigned short* kb = kT + (size_t)b * NP * C_;
    for (int kt = 0; kt < 65; ++kt) {
        const unsigned short* kp = kb + (size_t)(kt * 16 + ln) * C_;
        if (kt < 64)     // prefetch next key row (global_prefetch_b8 path)
            __builtin_prefetch(kp + C_ * 16, 0, 1);
        v8f s = {0.f, 0.f, 0.f, 0.f, 0.f, 0.f, 0.f, 0.f};
        #pragma unroll
        for (int ct = 0; ct < 8; ++ct) {
            AFrag kf;
            const unsigned short* p = kp + ct * 32 + coff;
            kf.q[0] = *(const u32x4*)p;
            kf.q[1] = *(const u32x4*)(p + 16);
            s = wmma_bf16(qf[ct], kf, s);
        }
        const int col = kt * 16 + ln;
        const bool bad = (col >= N_);
        #pragma unroll
        for (int r = 0; r < 8; ++r)
            E[(size_t)(r + half * 8) * NP + col] = bad ? NEG_INF : s[r];
    }
    // Fill padded tail columns 1040..1055 with -inf.
    #pragma unroll
    for (int j = 0; j < 8; ++j)
        E[(size_t)ln * NP + 1040 + half * 8 + j] = NEG_INF;
    asm volatile("s_wait_dscnt 0" ::: "memory");

    // ---- Phase 2: exact softmax stats; lane (ln,half) scans half a row ----
    float m = NEG_INF;
    {
        const float* er = E + (size_t)ln * NP + half * 528;
        for (int j = 0; j < 132; ++j) {
            f32x4 v4 = *(const f32x4*)(er + j * 4);
            m = fmaxf(m, fmaxf(fmaxf(v4[0], v4[1]), fmaxf(v4[2], v4[3])));
        }
    }
    m = fmaxf(m, __shfl_xor(m, 16, 32));
    float ssum = 0.f;
    {
        float* er = E + (size_t)ln * NP + half * 528;
        for (int j = 0; j < 132; ++j) {
            f32x4 v4 = *(f32x4*)(er + j * 4);
            v4[0] = __expf(v4[0] - m); v4[1] = __expf(v4[1] - m);
            v4[2] = __expf(v4[2] - m); v4[3] = __expf(v4[3] - m);
            ssum += v4[0] + v4[1] + v4[2] + v4[3];
            *(f32x4*)(er + j * 4) = v4;      // write P back in place
        }
    }
    ssum += __shfl_xor(ssum, 16, 32);
    const float inv = 1.0f / ssum;
    asm volatile("s_wait_dscnt 0" ::: "memory");

    // ---- Phase 3: outT(q,c) += P(q,k) * v^T(k,c) ----
    v8f oAcc[16];
    #pragma unroll
    for (int ct = 0; ct < 16; ++ct)
        #pragma unroll
        for (int r = 0; r < 8; ++r) oAcc[ct][r] = 0.f;

    const unsigned short* vb = vB + (size_t)b * C_ * NP;
    for (int kc = 0; kc < 33; ++kc) {
        const int kbase = kc * 32;
        // Build P A-fragment from LDS (lane owns row ln).
        AFrag pf;
        const float* er = E + (size_t)ln * NP + kbase;
        pf.b4[0] = cvt4(*(const f32x4*)(er + coff));
        pf.b4[1] = cvt4(*(const f32x4*)(er + coff + 4));
        pf.b4[2] = cvt4(*(const f32x4*)(er + 16 + coff));
        pf.b4[3] = cvt4(*(const f32x4*)(er + 16 + coff + 4));
        #pragma unroll
        for (int ct = 0; ct < 16; ++ct) {
            AFrag vf;   // B-fragment: rows of (v^T)^T = v(C,NP), lane col c
            const unsigned short* vp = vb + (size_t)(ct * 16 + ln) * NP + kbase + coff;
            vf.q[0] = *(const u32x4*)vp;
            vf.q[1] = *(const u32x4*)(vp + 16);
            oAcc[ct] = wmma_bf16(pf, vf, oAcc[ct]);
        }
    }

    // ---- Phase 4: y = gamma*x + O/rowsum, stored (B, C, NP) ----
    float invs[8];
    #pragma unroll
    for (int j = 0; j < 8; ++j) invs[j] = __shfl(inv, half * 8 + j, 32);
    const float g = gamma[0];
    float* yB = yW + (size_t)b * C_ * NP;
    const float* xB = x + (size_t)b * C_ * N_;

    #pragma unroll
    for (int ct = 0; ct < 16; ++ct) {
        const int c = ct * 16 + ln;
        if (q0 + 15 < N_) {
            f32x4 o0, o1;
            #pragma unroll
            for (int r = 0; r < 4; ++r) {
                o0[r] = oAcc[ct][r]     * invs[r]     + g * xB[(size_t)c * N_ + q0 + half * 8 + r];
                o1[r] = oAcc[ct][r + 4] * invs[r + 4] + g * xB[(size_t)c * N_ + q0 + half * 8 + r + 4];
            }
            float* yp = &yB[(size_t)c * NP + q0 + half * 8];
            *(f32x4*)yp       = o0;
            *(f32x4*)(yp + 4) = o1;
        } else {
            #pragma unroll
            for (int r = 0; r < 8; ++r) {
                const int q = q0 + half * 8 + r;
                if (q < N_)
                    yB[(size_t)c * NP + q] =
                        oAcc[ct][r] * invs[r] + g * xB[(size_t)c * N_ + q];
            }
        }
    }
}

// ---------------------------------------------------------------------------
// Kernel 3a: deterministic per-channel partial sums over (B, N).
// ---------------------------------------------------------------------------
__global__ void k_bnstat(const float* __restrict__ yW,
                         float* __restrict__ pS, float* __restrict__ pQ) {
    __shared__ float sh[256], sh2[256];
    const int c = blockIdx.x, s = blockIdx.y, t = threadIdx.x;
    const int TOT = B_ * N_;
    float sum = 0.f, sq = 0.f;
    for (int i = s * 256 + t; i < TOT; i += 16 * 256) {
        const int bb = i / N_;
        const int q  = i - bb * N_;
        const float yv = yW[(size_t)bb * C_ * NP + (size_t)c * NP + q];
        sum += yv; sq += yv * yv;
    }
    sh[t] = sum; sh2[t] = sq;
    __syncthreads();
    for (int ofs = 128; ofs > 0; ofs >>= 1) {
        if (t < ofs) { sh[t] += sh[t + ofs]; sh2[t] += sh2[t + ofs]; }
        __syncthreads();
    }
    if (t == 0) { pS[c * 16 + s] = sh[0]; pQ[c * 16 + s] = sh2[0]; }
}

// Kernel 3b: finalize BN stats into per-channel scale/shift.
__global__ void k_bnfinal(const float* __restrict__ pS, const float* __restrict__ pQ,
                          const float* __restrict__ bw, const float* __restrict__ bb,
                          float* __restrict__ scale, float* __restrict__ shift) {
    const int c = threadIdx.x;
    float s = 0.f, q = 0.f;
    for (int i = 0; i < 16; ++i) { s += pS[c * 16 + i]; q += pQ[c * 16 + i]; }
    const float invN = 1.0f / (float)(B_ * N_);
    const float mean = s * invN;
    const float var  = q * invN - mean * mean;
    const float istd = rsqrtf(var + 1e-5f);
    const float sc = istd * bw[c];
    scale[c] = sc;
    shift[c] = bb[c] - mean * sc;
}

// Kernel 4: normalize padded y into the exact (B, C, N) output.
__global__ void k_norm(const float* __restrict__ yW,
                       const float* __restrict__ scale,
                       const float* __restrict__ shift,
                       float* __restrict__ out) {
    const long long idx = (long long)blockIdx.x * blockDim.x + threadIdx.x;
    const long long TOT = (long long)B_ * C_ * N_;
    if (idx >= TOT) return;
    const int bb = (int)(idx / (C_ * N_));
    const int r  = (int)(idx - (long long)bb * C_ * N_);
    const int c  = r / N_;
    const int q  = r - c * N_;
    out[idx] = yW[(size_t)bb * C_ * NP + (size_t)c * NP + q] * scale[c] + shift[c];
}

// ---------------------------------------------------------------------------
extern "C" void kernel_launch(void* const* d_in, const int* in_sizes, int n_in,
                              void* d_out, int out_size, void* d_ws, size_t ws_size,
                              hipStream_t stream) {
    const float* x     = (const float*)d_in[0];
    const float* Wq    = (const float*)d_in[1];
    const float* bq    = (const float*)d_in[2];
    const float* Wk    = (const float*)d_in[3];
    const float* bk    = (const float*)d_in[4];
    const float* Wv    = (const float*)d_in[5];
    const float* bv    = (const float*)d_in[6];
    const float* freqs = (const float*)d_in[7];
    const float* gamma = (const float*)d_in[8];
    const float* bw    = (const float*)d_in[9];
    const float* bb    = (const float*)d_in[10];
    float* out = (float*)d_out;

    char* ws = (char*)d_ws;
    size_t off = 0;
    auto alloc = [&](size_t bytes) -> char* {
        char* p = ws + off;
        off += bytes;
        off = (off + 255) & ~(size_t)255;
        return p;
    };
    float*          cosT = (float*)alloc((size_t)1024 * 128 * 4);
    float*          sinT = (float*)alloc((size_t)1024 * 128 * 4);
    unsigned short* qT   = (unsigned short*)alloc((size_t)B_ * NP * C_ * 2);
    unsigned short* kT   = (unsigned short*)alloc((size_t)B_ * NP * C_ * 2);
    unsigned short* vB   = (unsigned short*)alloc((size_t)B_ * C_ * NP * 2);
    float*          yW   = (float*)alloc((size_t)B_ * C_ * NP * 4);
    float*          pS   = (float*)alloc((size_t)256 * 16 * 4);
    float*          pQ   = (float*)alloc((size_t)256 * 16 * 4);
    float*          scl  = (float*)alloc((size_t)256 * 4);
    float*          shf  = (float*)alloc((size_t)256 * 4);

    const int attnLds = 4 * 16 * NP * 4;   // 270336 B per block (<= 320KB/WGP)
    (void)hipFuncSetAttribute((const void*)k_attn,
                              hipFuncAttributeMaxDynamicSharedMemorySize, attnLds);

    k_tables<<<512, 256, 0, stream>>>(freqs, cosT, sinT);
    k_proj<<<dim3(66, 64), 128, 0, stream>>>(x, Wq, bq, Wk, bk, Wv, bv,
                                             cosT, sinT, qT, kT, vB);
    k_attn<<<dim3(64, 17), 128, attnLds, stream>>>(qT, kT, vB, x, gamma, yW);
    k_bnstat<<<dim3(256, 16), 256, 0, stream>>>(yW, pS, pQ);
    k_bnfinal<<<1, 256, 0, stream>>>(pS, pQ, bw, bb, scl, shf);
    k_norm<<<(B_ * C_ * N_ + 255) / 256, 256, 0, stream>>>(yW, scl, shf, out);
}
